// MSA2Pair_3307124817994
// MI455X (gfx1250) — compile-verified, matrix operand order checked by reading, not run on hardware
//
#include <hip/hip_runtime.h>

typedef __attribute__((ext_vector_type(16))) _Float16 v16h;
typedef __attribute__((ext_vector_type(8)))  _Float16 v8h;
typedef __attribute__((ext_vector_type(8)))  float    v8f;

#define L_RES   256
#define N_SEQ   128
#define D_MSA   256
#define D_HID   32
#define D_PAIR  128

#if defined(__AMDGCN__) && __has_builtin(__builtin_amdgcn_global_load_async_to_lds_b128)
#define HAVE_ASYNC_LDS 1
#else
#define HAVE_ASYNC_LDS 0
#endif

#if HAVE_ASYNC_LDS
// builtin prototype is (v4i addrspace(1)*, v4i addrspace(3)*, imm offset, imm cpol)
typedef __attribute__((__vector_size__(4 * sizeof(int)))) int vi4;
typedef __attribute__((address_space(1))) vi4 gvi4;
typedef __attribute__((address_space(3))) vi4 lvi4;
#endif

// ---------------------------------------------------------------------------
// 16-byte global -> LDS copy: async (ASYNCcnt-tracked, no VGPR round trip)
// on CDNA5 toolchains that expose the builtin, plain copy otherwise.
// ---------------------------------------------------------------------------
__device__ __forceinline__ void cp16(_Float16* dst_lds, const _Float16* src_g) {
#if HAVE_ASYNC_LDS
    __builtin_amdgcn_global_load_async_to_lds_b128(
        (gvi4*)(src_g), (lvi4*)(dst_lds), /*offset=*/0, /*cpol=*/0);
#else
    *(uint4*)dst_lds = *(const uint4*)src_g;
#endif
}

__device__ __forceinline__ void async_fence() {
#if HAVE_ASYNC_LDS
#if __has_builtin(__builtin_amdgcn_s_wait_asynccnt)
    __builtin_amdgcn_s_wait_asynccnt(0);
#else
    asm volatile("s_wait_asynccnt 0" ::: "memory");
#endif
#endif
}

// ---------------------------------------------------------------------------
// fragment loader: halves [0..7] from p, [8..15] from p+16 (K-contiguous rows)
// matches CDNA5 16-bit A/B layout: lane<16 -> K {kb..kb+7, kb+16..kb+23},
// lane>=16 -> kb has +8 applied by caller.
// ---------------------------------------------------------------------------
__device__ __forceinline__ v16h load_frag(const _Float16* p) {
    v16h r;
    ((uint4*)&r)[0] = *(const uint4*)(p);
    ((uint4*)&r)[1] = *(const uint4*)(p + 16);
    return r;
}

__device__ __forceinline__ v8f wmma_f16(v16h a, v16h b, v8f c) {
    return __builtin_amdgcn_wmma_f32_16x16x32_f16(
        /*neg_a=*/false, a, /*neg_b=*/false, b,
        /*c_mod=*/(short)0, c, /*reuse_a=*/false, /*reuse_b=*/false);
}

// ---------------------------------------------------------------------------
// prep: pack [w_left | w_right/N] into f16 [64 n][256 k] (K-contiguous rows)
// ---------------------------------------------------------------------------
__global__ void prep_wcat(const float* __restrict__ wl, const float* __restrict__ bl,
                          const float* __restrict__ wr, const float* __restrict__ br,
                          _Float16* __restrict__ wcat, float* __restrict__ bcat) {
    int idx = blockIdx.x * 256 + threadIdx.x;       // 64*256 = 16384
    if (idx < 64 * 256) {
        int n = idx >> 8, k = idx & 255;
        float v = (n < 32) ? wl[k * 32 + n]
                           : wr[k * 32 + (n - 32)] * (1.0f / (float)N_SEQ);
        wcat[n * 256 + k] = (_Float16)v;
    }
    if (idx < 64) {
        bcat[idx] = (idx < 32) ? bl[idx] : br[idx - 32] * (1.0f / (float)N_SEQ);
    }
}

// prep: w_out [1024 ij][128 p] -> f16 w_out_t [128 p][1024 ij]
__global__ void prep_wout(const float* __restrict__ wout, _Float16* __restrict__ wout_t) {
    int idx = blockIdx.x * 256 + threadIdx.x;       // 1024*128 = 131072
    int ij = idx >> 7, p = idx & 127;
    wout_t[p * 1024 + ij] = (_Float16)wout[idx];
}

// ---------------------------------------------------------------------------
// Stage 1: LayerNorm + dual projection.  One WG = 64 rows (row = s*L + l).
// GEMM [64 x 256] @ [256 x 64] via WMMA; results scattered transposed:
//   left_t[(l*32+i)][s], right_t[(l*32+j)][s]   (f16, K=s contiguous)
// ---------------------------------------------------------------------------
__global__ __launch_bounds__(256) void msa_ln_proj(
    const float* __restrict__ msa, const float* __restrict__ gamma,
    const float* __restrict__ beta, const _Float16* __restrict__ wcat,
    const float* __restrict__ bcat,
    _Float16* __restrict__ left_t, _Float16* __restrict__ right_t) {

    __shared__ _Float16 sX[64 * 264];   // 64 rows x (256 + 8 pad)
    __shared__ _Float16 sW[64 * 264];

    const int tid  = threadIdx.x;
    const int lane = tid & 31;
    const int wave = tid >> 5;
    const int r0   = blockIdx.x * 64;

    // ---- stage packed weights into LDS (B operand, K-contiguous rows) ----
    {
        int row = tid >> 2, q = tid & 3;            // 4 threads/row, 64 f16 each
#pragma unroll
        for (int u = 0; u < 8; ++u)
            cp16(sW + row * 264 + q * 64 + u * 8,
                 wcat + row * 256 + q * 64 + u * 8);
    }

    // ---- LayerNorm: wave handles 8 rows, lane handles 8 elements ----
    const int d = lane * 8;
    float4 g0 = ((const float4*)(gamma + d))[0];
    float4 g1 = ((const float4*)(gamma + d))[1];
    float4 e0 = ((const float4*)(beta  + d))[0];
    float4 e1 = ((const float4*)(beta  + d))[1];
    float gg[8] = {g0.x, g0.y, g0.z, g0.w, g1.x, g1.y, g1.z, g1.w};
    float bb[8] = {e0.x, e0.y, e0.z, e0.w, e1.x, e1.y, e1.z, e1.w};

    for (int rr = 0; rr < 8; ++rr) {
        int row_local = wave * 8 + rr;
        const float4* p4 = (const float4*)(msa + (size_t)(r0 + row_local) * D_MSA + d);
        float4 a0 = p4[0], a1 = p4[1];
        float xs[8] = {a0.x, a0.y, a0.z, a0.w, a1.x, a1.y, a1.z, a1.w};
        float s = 0.f, ss = 0.f;
#pragma unroll
        for (int u = 0; u < 8; ++u) { s += xs[u]; ss += xs[u] * xs[u]; }
#pragma unroll
        for (int off = 16; off > 0; off >>= 1) {
            s  += __shfl_xor(s,  off, 32);
            ss += __shfl_xor(ss, off, 32);
        }
        float mu  = s * (1.0f / 256.0f);
        float var = ss * (1.0f / 256.0f) - mu * mu;
        float inv = rsqrtf(var + 1e-5f);
        v8h o;
#pragma unroll
        for (int u = 0; u < 8; ++u)
            o[u] = (_Float16)((xs[u] - mu) * inv * gg[u] + bb[u]);
        *(v8h*)(sX + row_local * 264 + d) = o;
    }
    async_fence();
    __syncthreads();

    // ---- WMMA: wave -> M-tile (wave&3), 2 N-tiles ((wave>>2)*2 ..+1) ----
    const int mt   = wave & 3;
    const int ntb  = (wave >> 2) * 2;
    const int nc   = lane & 15;
    const int koff = ((lane >> 4) & 1) * 8;
    const int mofs = (lane >> 4) * 8;
    const int arow = mt * 16 + nc;

    v8f acc0 = {}, acc1 = {};
    for (int kt = 0; kt < 8; ++kt) {
        int kb = kt * 32 + koff;
        v16h af = load_frag(sX + arow * 264 + kb);
        v16h b0 = load_frag(sW + (ntb * 16 + nc) * 264 + kb);
        v16h b1 = load_frag(sW + ((ntb + 1) * 16 + nc) * 264 + kb);
        acc0 = wmma_f16(af, b0, acc0);
        acc1 = wmma_f16(af, b1, acc1);
    }

    // ---- epilogue: +bias, convert f16, scatter transposed ----
#pragma unroll
    for (int t = 0; t < 2; ++t) {
        v8f acc = t ? acc1 : acc0;
        int n = (ntb + t) * 16 + nc;
        float bias = bcat[n];
#pragma unroll
        for (int v = 0; v < 8; ++v) {
            int row_local = mt * 16 + mofs + v;
            int r = r0 + row_local;
            int sidx = r >> 8;          // s
            int li   = r & 255;         // l
            _Float16 h = (_Float16)(acc[v] + bias);
            if (n < 32) left_t [(size_t)(li * 32 + n)        * 128 + sidx] = h;
            else        right_t[(size_t)(li * 32 + (n - 32)) * 128 + sidx] = h;
        }
    }
}

// ---------------------------------------------------------------------------
// Stage 2+3 fused.  WG tile: 4 l x 16 m = 64 (l,m) pairs.
//   stage2: O[(l,i)][(m,j)] = sum_s left*right   (128 x 512, K=128) -> LDS f16
//   stage3: out[q=(l,m)][p] = O[q][:1024] @ w_out_t^T + b_out + pair
// ---------------------------------------------------------------------------
__global__ __launch_bounds__(256) void outer_proj(
    const _Float16* __restrict__ left_t, const _Float16* __restrict__ right_t,
    const _Float16* __restrict__ wout_t, const float* __restrict__ bout,
    const float* __restrict__ pair, float* __restrict__ out) {

    __shared__ _Float16 smem[128 * 136 + 512 * 136 + 64 * 1032]; // 299 KB
    _Float16* sA = smem;                         // 128 rows (l,i) x (128+8)
    _Float16* sB = smem + 128 * 136;             // 512 rows (m,j) x (128+8)
    _Float16* sO = smem + 128 * 136 + 512 * 136; // 64 rows q x (1024+8)

    const int tid  = threadIdx.x;
    const int lane = tid & 31;
    const int wave = tid >> 5;
    const int lblk = blockIdx.x >> 4;
    const int mblk = blockIdx.x & 15;
    const int l0   = lblk * 4;
    const int m0   = mblk * 16;

    // ---- stage A/B panels into LDS (async on CDNA5) ----
    {
        int row = tid >> 1, h = tid & 1;         // 128 rows, 2 thr/row
#pragma unroll
        for (int u = 0; u < 8; ++u)
            cp16(sA + row * 136 + h * 64 + u * 8,
                 left_t + (size_t)(l0 * 32 + row) * 128 + h * 64 + u * 8);
    }
#pragma unroll
    for (int rr = 0; rr < 2; ++rr) {             // 512 rows, 2 rows/thr
        int row = tid * 2 + rr;
#pragma unroll
        for (int u = 0; u < 16; ++u)
            cp16(sB + row * 136 + u * 8,
                 right_t + (size_t)(m0 * 32 + row) * 128 + u * 8);
    }
    async_fence();
    __syncthreads();

    const int nc   = lane & 15;
    const int koff = ((lane >> 4) & 1) * 8;
    const int mofs = (lane >> 4) * 8;

    // ---- stage 2: wave owns M-strip `wave`, sweeps 32 N-tiles in groups of 4
    {
        const int mt   = wave;
        const int arow = mt * 16 + nc;
        // hoist the 4 A fragments (reused by all 32 N-tiles)
        v16h afr[4];
#pragma unroll
        for (int kt = 0; kt < 4; ++kt)
            afr[kt] = load_frag(sA + arow * 136 + kt * 32 + koff);

        for (int ng = 0; ng < 8; ++ng) {
            v8f z = {};
            v8f acc[4] = {z, z, z, z};
#pragma unroll
            for (int kt = 0; kt < 4; ++kt) {
                int kb = kt * 32 + koff;
#pragma unroll
                for (int u = 0; u < 4; ++u) {
                    int nt = ng * 4 + u;
                    v16h bf = load_frag(sB + (nt * 16 + nc) * 136 + kb);
                    acc[u] = wmma_f16(afr[kt], bf, acc[u]);
                }
            }
            // transpose-scatter C-layout -> sO[q=(l,m)][ij] (f16)
#pragma unroll
            for (int u = 0; u < 4; ++u) {
                int nt = ng * 4 + u;
                int c2 = nt * 16 + nc;                  // (m_local, j)
                int mloc = c2 >> 5, j = c2 & 31;
#pragma unroll
                for (int v = 0; v < 8; ++v) {
                    int r2 = mt * 16 + mofs + v;        // (l_local, i)
                    int lloc = r2 >> 5, i = r2 & 31;
                    sO[(lloc * 16 + mloc) * 1032 + i * 32 + j] = (_Float16)acc[u][v];
                }
            }
        }
    }
    __syncthreads();

    // ---- stage 3: [64 x 1024] @ w_out_t (K=1024), +b_out +pair ----
    {
        const int mt3   = wave & 3;                 // q-strip
        const int ntb3  = (wave >> 2) * 4;          // 4 p-tiles
        const int arow3 = mt3 * 16 + nc;
        v8f z = {};
        v8f acc[4] = {z, z, z, z};
        for (int kt = 0; kt < 32; ++kt) {
            int kb = kt * 32 + koff;
            v16h af = load_frag(sO + arow3 * 1032 + kb);
#pragma unroll
            for (int u = 0; u < 4; ++u) {
                int p = (ntb3 + u) * 16 + nc;
                v16h bf = load_frag(wout_t + (size_t)p * 1024 + kb);
                acc[u] = wmma_f16(af, bf, acc[u]);
            }
        }
#pragma unroll
        for (int u = 0; u < 4; ++u) {
            int p = (ntb3 + u) * 16 + nc;
            float bo = bout[p];
#pragma unroll
            for (int v = 0; v < 8; ++v) {
                int q = mt3 * 16 + mofs + v;
                int lloc = q >> 4, mloc = q & 15;
                size_t oi = ((size_t)((l0 + lloc) * L_RES + (m0 + mloc))) * D_PAIR + p;
                out[oi] = acc[u][v] + bo + pair[oi];
            }
        }
    }
}

// ---------------------------------------------------------------------------
extern "C" void kernel_launch(void* const* d_in, const int* in_sizes, int n_in,
                              void* d_out, int out_size, void* d_ws, size_t ws_size,
                              hipStream_t stream) {
    const float* msa     = (const float*)d_in[0];
    const float* pair    = (const float*)d_in[1];
    // d_in[2]=symmids, d_in[3]=symmsub : unused by reference
    const float* gamma   = (const float*)d_in[4];
    const float* beta    = (const float*)d_in[5];
    const float* w_left  = (const float*)d_in[6];
    const float* b_left  = (const float*)d_in[7];
    const float* w_right = (const float*)d_in[8];
    const float* b_right = (const float*)d_in[9];
    const float* w_out   = (const float*)d_in[10];
    const float* b_out   = (const float*)d_in[11];
    float* out = (float*)d_out;

    char* ws = (char*)d_ws;
    _Float16* left_t  = (_Float16*)(ws);                       // 8192*128 f16 = 2 MB
    _Float16* right_t = (_Float16*)(ws + (2u << 20));          // 2 MB
    _Float16* wcat    = (_Float16*)(ws + (4u << 20));          // 64*256 f16 = 32 KB
    float*    bcat    = (float*)   (ws + (4u << 20) + 32768);  // 64 f32
    _Float16* wout_t  = (_Float16*)(ws + (4u << 20) + 65536);  // 128*1024 f16 = 256 KB

    prep_wcat <<<64,  256, 0, stream>>>(w_left, b_left, w_right, b_right, wcat, bcat);
    prep_wout <<<512, 256, 0, stream>>>(w_out, wout_t);
    msa_ln_proj<<<512, 256, 0, stream>>>(msa, gamma, beta, wcat, bcat, left_t, right_t);
    outer_proj<<<1024, 256, 0, stream>>>(left_t, right_t, wout_t, b_out, pair, out);
}
